// MSADF_DTA_46986942218283
// MI455X (gfx1250) — matrix operand order repaired and checked by reading, not compiled, not run on hardware
//
#include <hip/hip_runtime.h>
#include <hip/hip_bf16.h>

typedef __attribute__((ext_vector_type(16))) __bf16 v16bf;
typedef __attribute__((ext_vector_type(8)))  __bf16 v8bf;
typedef __attribute__((ext_vector_type(8)))  float  v8f;

#define GRID_STRIDE(i, total) \
  for (long i = (long)blockIdx.x * blockDim.x + threadIdx.x; i < (total); i += (long)gridDim.x * blockDim.x)

// ---------------------------------------------------------------------------
// Utility kernels
// ---------------------------------------------------------------------------
__global__ void k_zero(float* p, long n) {
  GRID_STRIDE(i, n) p[i] = 0.f;
}

__global__ void k_zero2d(float* p, long ld, long N, int C) {
  long total = N * (long)C;
  GRID_STRIDE(i, total) {
    long n = i / C; int c = (int)(i - n * C);
    p[n * ld + c] = 0.f;
  }
}

// Copy [N,cin] row-major into [N,ld] with zero padding
__global__ void k_pad_copy(const float* __restrict__ x, int cin, float* __restrict__ out,
                           int ld, long N) {
  long total = N * (long)ld;
  GRID_STRIDE(i, total) {
    long n = i / ld; int c = (int)(i - n * ld);
    out[i] = (c < cin) ? x[n * cin + c] : 0.f;
  }
}

// Repack weight [K,N] row-major -> transposed padded bf16 Wt [N][Kpad] (zero pad)
__global__ void k_repackT(const float* __restrict__ Wsrc, __bf16* __restrict__ Wt,
                          int K, int N, int Kp) {
  long total = (long)N * Kp;
  GRID_STRIDE(i, total) {
    long n = i / Kp; int k = (int)(i - n * Kp);
    Wt[i] = (__bf16)((k < K) ? Wsrc[(long)k * N + n] : 0.f);
  }
}

// Conv weight w [96, Cin, 3] -> bf16 Wt [96][3*Cin] with Wt[o][k*Cin+ci] = w[o][ci][k]
__global__ void k_repack_convT(const float* __restrict__ w, __bf16* __restrict__ Wt, int Cin) {
  long total = 96L * 3 * Cin;
  GRID_STRIDE(i, total) {
    long o = i / (3 * Cin);
    int r = (int)(i - o * 3 * Cin);
    int k = r / Cin, ci = r - (r / Cin) * Cin;
    Wt[i] = (__bf16)w[((long)o * Cin + ci) * 3 + k];
  }
}

// agg[dst] += x[src], 4 channels per thread
__global__ void k_edge_agg4(const float* __restrict__ x, long ldx,
                            const int* __restrict__ src, const int* __restrict__ dst,
                            float* __restrict__ agg, long ldagg, long E, int C4) {
  long total = E * (long)C4;
  GRID_STRIDE(i, total) {
    long e = i / C4; int c4 = (int)(i - e * C4);
    const float4 v = *(const float4*)(x + (long)src[e] * ldx + c4 * 4);
    float* d = agg + (long)dst[e] * ldagg + c4 * 4;
    atomicAdd(d + 0, v.x); atomicAdd(d + 1, v.y);
    atomicAdd(d + 2, v.z); atomicAdd(d + 3, v.w);
  }
}

__global__ void k_bn_stats(const float* __restrict__ h, long ld, int N, int C,
                           float* __restrict__ mean, float* __restrict__ var) {
  int c = blockIdx.x;
  if (c >= C) return;
  float s = 0.f, sq = 0.f;
  for (int n = threadIdx.x; n < N; n += blockDim.x) {
    float v = h[(long)n * ld + c];
    s += v; sq += v * v;
  }
  __shared__ float ss[256], s2[256];
  ss[threadIdx.x] = s; s2[threadIdx.x] = sq;
  __syncthreads();
  for (int off = blockDim.x >> 1; off > 0; off >>= 1) {
    if ((int)threadIdx.x < off) { ss[threadIdx.x] += ss[threadIdx.x + off]; s2[threadIdx.x] += s2[threadIdx.x + off]; }
    __syncthreads();
  }
  if (threadIdx.x == 0) {
    float mu = ss[0] / (float)N;
    mean[c] = mu;
    var[c] = s2[0] / (float)N - mu * mu;
  }
}

__global__ void k_bn_relu(const float* __restrict__ h, long ldh, float* __restrict__ out, long ldo,
                          const float* __restrict__ mean, const float* __restrict__ var,
                          const float* __restrict__ g, const float* __restrict__ be,
                          long N, int C) {
  long total = N * (long)C;
  GRID_STRIDE(i, total) {
    long n = i / C; int c = (int)(i - n * C);
    float v = (h[n * ldh + c] - mean[c]) * rsqrtf(var[c] + 1e-5f) * g[c] + be[c];
    out[n * ldo + c] = v > 0.f ? v : 0.f;
  }
}

__global__ void k_pool_accum4(const float* __restrict__ x, long ldx, const int* __restrict__ bid,
                              float* __restrict__ sums, long lds, long N, int C4) {
  long total = N * (long)C4;
  GRID_STRIDE(i, total) {
    long n = i / C4; int c4 = (int)(i - n * C4);
    const float4 v = *(const float4*)(x + n * ldx + c4 * 4);
    float* d = sums + (long)bid[n] * lds + c4 * 4;
    atomicAdd(d + 0, v.x); atomicAdd(d + 1, v.y);
    atomicAdd(d + 2, v.z); atomicAdd(d + 3, v.w);
  }
}

__global__ void k_pool_cnt(const int* __restrict__ bid, float* __restrict__ cnt, long N) {
  GRID_STRIDE(i, N) atomicAdd(&cnt[bid[i]], 1.f);
}

__global__ void k_pool_div(float* sums, long lds, const float* cnt, int Bn, int C) {
  long total = (long)Bn * C;
  GRID_STRIDE(i, total) {
    long b = i / C; int c = (int)(i - b * C);
    sums[b * lds + c] /= fmaxf(cnt[b], 1.f);
  }
}

// Protein embedding -> channels-last padded [B, L+2, 128]
__global__ void k_embed(const int* __restrict__ tok, const float* __restrict__ Ew,
                        float* __restrict__ xpad, int Bn, int L) {
  long total = (long)Bn * L * 128;
  GRID_STRIDE(i, total) {
    long b = i / ((long)L * 128);
    long rem = i - b * (long)L * 128;
    int l = (int)(rem / 128), c = (int)(rem - (long)(rem / 128) * 128);
    int t = tok[b * L + l];
    float v = (t == 0) ? 0.f : Ew[t * 128 + c];
    xpad[(b * (L + 2) + l + 1) * 128 + c] = v;
  }
}

__global__ void k_maxpool(const float* __restrict__ x, int Bn, int L,
                          float* __restrict__ out, long ldo, int colOff) {
  int i = blockIdx.x * blockDim.x + threadIdx.x;
  if (i >= Bn * 96) return;
  int b = i / 96, o = i - (i / 96) * 96;
  float m = -3.4e38f;
  for (int l = 0; l < L; ++l)
    m = fmaxf(m, x[((long)b * (L + 2) + l + 1) * 128 + o]);
  out[(long)b * ldo + colOff + o] = m;
}

__global__ void k_concat2(const float* __restrict__ a, const float* __restrict__ b,
                          float* __restrict__ out, int Bn, int C) {
  long total = (long)Bn * 2 * C;
  GRID_STRIDE(i, total) {
    long n = i / (2 * C); int c = (int)(i - n * 2 * C);
    out[i] = (c < C) ? a[n * C + c] : b[n * C + c - C];
  }
}

__global__ void k_gate_fuse(const float* __restrict__ gate, const float* __restrict__ a,
                            const float* __restrict__ b, float* __restrict__ fin, int Bn) {
  long total = (long)Bn * 96;
  GRID_STRIDE(i, total) {
    long n = i / 96; int c = (int)(i - n * 96);
    float g = gate[i], av = a[i], bv = b[i];
    fin[n * 192 + c] = g * av + (1.f - g) * bv;
    fin[n * 192 + 96 + c] = av - bv;
  }
}

// ---------------------------------------------------------------------------
// bf16 helpers
// ---------------------------------------------------------------------------
__device__ __forceinline__ v16bf packA16(float4 a0, float4 a1, float4 a2, float4 a3) {
  v16bf v;
  v[0]  = (__bf16)a0.x; v[1]  = (__bf16)a0.y; v[2]  = (__bf16)a0.z; v[3]  = (__bf16)a0.w;
  v[4]  = (__bf16)a1.x; v[5]  = (__bf16)a1.y; v[6]  = (__bf16)a1.z; v[7]  = (__bf16)a1.w;
  v[8]  = (__bf16)a2.x; v[9]  = (__bf16)a2.y; v[10] = (__bf16)a2.z; v[11] = (__bf16)a2.w;
  v[12] = (__bf16)a3.x; v[13] = (__bf16)a3.y; v[14] = (__bf16)a3.z; v[15] = (__bf16)a3.w;
  return v;
}

__device__ __forceinline__ v16bf loadB16(const __bf16* p) {
  v8bf lo = *(const v8bf*)(p);
  v8bf hi = *(const v8bf*)(p + 16);
  return __builtin_shufflevector(lo, hi, 0, 1, 2, 3, 4, 5, 6, 7, 8, 9, 10, 11, 12, 13, 14, 15);
}

// ---------------------------------------------------------------------------
// WMMA GEMM: Out[M,N] = act( A[M,Kp] @ Bt^T + bias (+ Out if ACC) )
// A f32 row-major (lda mult of 4, readable/zero to Kp). Bt bf16 transposed
// padded [N][Kp]. Each wave: one 16x32 output tile (two WMMAs / k-step).
// ACT: 0 none, 1 relu, 2 sigmoid
// ---------------------------------------------------------------------------
template <int ACC, int ACT>
__global__ void k_gemm(const float* __restrict__ A, long lda,
                       const __bf16* __restrict__ Bt, long ldbt,
                       const float* __restrict__ bias,
                       float* __restrict__ Out, long ldo,
                       int M, int N, int Kp) {
  int lane = threadIdx.x & 31;
  int half = lane >> 4, lid = lane & 15;
  long row0 = (long)blockIdx.y * 16, col0 = (long)blockIdx.x * 32;
  long ar = row0 + lid;       if (ar >= M) ar = M - 1;   // clamp; stores guarded
  long bc0 = col0 + lid;      if (bc0 >= N) bc0 = N - 1;
  long bc1 = col0 + 16 + lid; if (bc1 >= N) bc1 = N - 1;
  const float*  Ar  = A  + ar  * lda;
  const __bf16* Br0 = Bt + bc0 * ldbt;
  const __bf16* Br1 = Bt + bc1 * ldbt;

  v8f acc0 = {0.f, 0.f, 0.f, 0.f, 0.f, 0.f, 0.f, 0.f};
  v8f acc1 = {0.f, 0.f, 0.f, 0.f, 0.f, 0.f, 0.f, 0.f};
  for (int k0 = 0; k0 < Kp; k0 += 32) {
    const float* Ap = Ar + k0 + half * 8;
    float4 a0 = *(const float4*)(Ap);
    float4 a1 = *(const float4*)(Ap + 4);
    float4 a2 = *(const float4*)(Ap + 16);
    float4 a3 = *(const float4*)(Ap + 20);
    v16bf bv0 = loadB16(Br0 + k0 + half * 8);
    v16bf bv1 = loadB16(Br1 + k0 + half * 8);
    v16bf av = packA16(a0, a1, a2, a3);
    acc0 = __builtin_amdgcn_wmma_f32_16x16x32_bf16(false, av, false, bv0, (short)0, acc0, false, false);
    acc1 = __builtin_amdgcn_wmma_f32_16x16x32_bf16(false, av, false, bv1, (short)0, acc1, false, false);
  }
  long c0 = col0 + lid, c1 = col0 + 16 + lid;
#pragma unroll
  for (int j = 0; j < 8; ++j) {
    long rr = row0 + j + 8 * half;
    if (rr < M) {
      float* orow = Out + rr * ldo;
      if (c0 < N) {
        float v = acc0[j];
        if (bias) v += bias[c0];
        if (ACC) v += orow[c0];
        if (ACT == 1) v = v > 0.f ? v : 0.f;
        if (ACT == 2) v = 1.f / (1.f + __expf(-v));
        orow[c0] = v;
      }
      if (c1 < N) {
        float v = acc1[j];
        if (bias) v += bias[c1];
        if (ACC) v += orow[c1];
        if (ACT == 1) v = v > 0.f ? v : 0.f;
        if (ACT == 2) v = 1.f / (1.f + __expf(-v));
        orow[c1] = v;
      }
    }
  }
}

// ---------------------------------------------------------------------------
// Implicit-im2col WMMA GEMM for conv1d (K=3, pad=1), channels-last padded X.
// X: [Bn, L+2, 128] (pad rows zero). Wt: bf16 [96][3*Cin]. Cin % 32 == 0 so
// each 32-chunk lies in a single kernel tap (kk uniform). 16x32 tile per wave.
// ---------------------------------------------------------------------------
__global__ void k_conv_gemm(const float* __restrict__ X, const __bf16* __restrict__ Wt,
                            const float* __restrict__ bias, float* __restrict__ Out,
                            int Bn, int L, int Cin) {
  const int M = Bn * L, K = 3 * Cin;
  const int Lp = L + 2, ld = 128;
  int lane = threadIdx.x & 31;
  int half = lane >> 4, lid = lane & 15;
  int row0 = blockIdx.y * 16, col0 = blockIdx.x * 32;
  int cc0 = col0 + lid, cc1 = col0 + 16 + lid;   // < 96 always (96 = 3*32)
  int m0 = row0 + lid; if (m0 >= M) m0 = M - 1;  // clamp; stores guarded
  int bb = m0 / L, l = m0 - bb * L;
  const float*  Xr  = X + ((long)bb * Lp + l) * ld;
  const __bf16* Br0 = Wt + (long)cc0 * K;
  const __bf16* Br1 = Wt + (long)cc1 * K;

  v8f acc0 = {0.f, 0.f, 0.f, 0.f, 0.f, 0.f, 0.f, 0.f};
  v8f acc1 = {0.f, 0.f, 0.f, 0.f, 0.f, 0.f, 0.f, 0.f};
  for (int k0 = 0; k0 < K; k0 += 32) {
    int kk = k0 / Cin;            // uniform across wave
    int ci0 = k0 - kk * Cin;
    const float* Ap = Xr + (long)kk * ld + ci0 + half * 8;
    float4 a0 = *(const float4*)(Ap);
    float4 a1 = *(const float4*)(Ap + 4);
    float4 a2 = *(const float4*)(Ap + 16);
    float4 a3 = *(const float4*)(Ap + 20);
    v16bf bv0 = loadB16(Br0 + k0 + half * 8);
    v16bf bv1 = loadB16(Br1 + k0 + half * 8);
    v16bf av = packA16(a0, a1, a2, a3);
    acc0 = __builtin_amdgcn_wmma_f32_16x16x32_bf16(false, av, false, bv0, (short)0, acc0, false, false);
    acc1 = __builtin_amdgcn_wmma_f32_16x16x32_bf16(false, av, false, bv1, (short)0, acc1, false, false);
  }
  // epilogue: one division, then increment-and-wrap through the 8 rows
  int mbase = row0 + 8 * half;
  int bb2 = mbase / L, l2 = mbase - (mbase / L) * L;
#pragma unroll
  for (int j = 0; j < 8; ++j) {
    int m = mbase + j;
    if (m < M) {
      float* orow = Out + ((long)bb2 * Lp + l2 + 1) * ld;
      float v0 = acc0[j] + bias[cc0]; orow[cc0] = v0 > 0.f ? v0 : 0.f;
      float v1 = acc1[j] + bias[cc1]; orow[cc1] = v1 > 0.f ? v1 : 0.f;
    }
    if (++l2 == L) { l2 = 0; ++bb2; }
  }
}

// ---------------------------------------------------------------------------
// Host-side orchestration
// ---------------------------------------------------------------------------
struct Gcbn { const __bf16 *WrT, *WsT; const float *b, *be, *g; int Kp, cout; };
struct Enc {
  Gcbn conv0, trans1, trans2;
  Gcbn b1c1[8], b1c2[8], b2c1[8], b2c2[8];
  const __bf16 *cWT;
  const float *cb;
  int cKp, nl;
};

static void gemm_f(hipStream_t s, int acc, int act,
                   const float* A, long lda, const __bf16* Bt, long ldbt, const float* bias,
                   float* Out, long ldo, int M, int N, int Kp) {
  dim3 g((N + 31) / 32, (M + 15) / 16), blk(32);
  if (acc == 0 && act == 0)      k_gemm<0, 0><<<g, blk, 0, s>>>(A, lda, Bt, ldbt, bias, Out, ldo, M, N, Kp);
  else if (acc == 1 && act == 0) k_gemm<1, 0><<<g, blk, 0, s>>>(A, lda, Bt, ldbt, bias, Out, ldo, M, N, Kp);
  else if (acc == 0 && act == 1) k_gemm<0, 1><<<g, blk, 0, s>>>(A, lda, Bt, ldbt, bias, Out, ldo, M, N, Kp);
  else                           k_gemm<0, 2><<<g, blk, 0, s>>>(A, lda, Bt, ldbt, bias, Out, ldo, M, N, Kp);
}

static void gcbn_fwd(hipStream_t s, const float* x, long ldx, long N,
                     const int* src, const int* dst, long E, const Gcbn& p,
                     float* out, long ldo, float* agg, long ldagg,
                     float* h, long ldh, float* stats) {
  k_zero2d<<<2048, 256, 0, s>>>(agg, ldagg, N, p.Kp);
  k_edge_agg4<<<4096, 256, 0, s>>>(x, ldx, src, dst, agg, ldagg, E, p.Kp / 4);
  gemm_f(s, 0, 0, x, ldx, p.WsT, p.Kp, p.b, h, ldh, (int)N, p.cout, p.Kp);
  gemm_f(s, 1, 0, agg, ldagg, p.WrT, p.Kp, nullptr, h, ldh, (int)N, p.cout, p.Kp);
  k_bn_stats<<<p.cout, 256, 0, s>>>(h, ldh, (int)N, p.cout, stats, stats + 256);
  k_bn_relu<<<2048, 256, 0, s>>>(h, ldh, out, ldo, stats, stats + 256, p.g, p.be, N, p.cout);
}

static void run_encoder(hipStream_t s, const float* x0, long ldx0, long N,
                        const int* src, const int* dst, long E, const int* bid,
                        const Enc& e, float* xcat, long ldx, float* agg, long ldagg,
                        float* h, long ldh, float* thid, long ldt, float* stats,
                        float* poolsum, float* poolcnt, float* outBuf) {
  gcbn_fwd(s, x0, ldx0, N, src, dst, E, e.conv0, xcat, ldx, agg, ldagg, h, ldh, stats);
  int nf = 32;
  for (int blk = 0; blk < 2; ++blk) {
    const Gcbn* c1s = (blk == 0) ? e.b1c1 : e.b2c1;
    const Gcbn* c2s = (blk == 0) ? e.b1c2 : e.b2c2;
    for (int li = 0; li < e.nl; ++li) {
      gcbn_fwd(s, xcat, ldx, N, src, dst, E, c1s[li], thid, ldt, agg, ldagg, h, ldh, stats);
      gcbn_fwd(s, thid, ldt, N, src, dst, E, c2s[li], xcat + nf + li * 32, ldx, agg, ldagg, h, ldh, stats);
    }
    nf += e.nl * 32;
    const Gcbn& tr = (blk == 0) ? e.trans1 : e.trans2;
    gcbn_fwd(s, xcat, ldx, N, src, dst, E, tr, xcat, ldx, agg, ldagg, h, ldh, stats);
    nf = tr.cout;
  }
  int C = nf;
  k_zero<<<256, 256, 0, s>>>(poolsum, 256L * 256);
  k_zero<<<1, 256, 0, s>>>(poolcnt, 256);
  k_pool_accum4<<<2048, 256, 0, s>>>(xcat, ldx, bid, poolsum, 256, N, C / 4);
  k_pool_cnt<<<512, 256, 0, s>>>(bid, poolcnt, N);
  k_pool_div<<<256, 256, 0, s>>>(poolsum, 256, poolcnt, 256, C);
  gemm_f(s, 0, 0, poolsum, 256, e.cWT, e.cKp, e.cb, outBuf, 96, 256, 96, e.cKp);
}

extern "C" void kernel_launch(void* const* d_in, const int* in_sizes, int n_in,
                              void* d_out, int out_size, void* d_ws, size_t ws_size,
                              hipStream_t stream) {
  (void)n_in; (void)out_size; (void)ws_size;
  const float* atomic_x = (const float*)d_in[0];
  const float* brics_x  = (const float*)d_in[1];
  const float* P        = (const float*)d_in[2];
  const int*   aei      = (const int*)d_in[3];
  const int*   bei      = (const int*)d_in[4];
  const int*   abid     = (const int*)d_in[5];
  const int*   bbid     = (const int*)d_in[6];
  const int*   target   = (const int*)d_in[7];

  const long NA = 65536, NB = 8192, Bn = 256, L = 1000;
  const long EA = in_sizes[3] / 2, EB = in_sizes[4] / 2;
  const int* asrc = aei;  const int* adst = aei + EA;
  const int* bsrc = bei;  const int* bdst = bei + EB;

  // ---- Workspace layout (floats) ----
  float* W = (float*)d_ws;
  float* AX  = W;                          // atomic xcat [65536,512] / protein pbufA
  float* AG  = W + 33554432L;              // atomic agg  [65536,512] / protein pbufB
  float* AH  = W + 67108864L;              // atomic h [65536,256]; later brics region
  float* BX  = AH;                         // brics xcat [8192,256]
  float* BG  = W + 67108864L + 2097152L;   // brics agg [8192,768]
  float* BHh = W + 67108864L + 8388608L;   // brics h [8192,128]
  float* TH  = W + 83886080L;              // padded atomic_x [65536,32] / thid [65536,64]
  float* S   = W + 88080384L;
  float* stats   = S;                      // 512
  float* poolsum = S + 512;                // 65536
  float* poolcnt = poolsum + 65536;        // 256
  float* abuf    = poolcnt + 256;          // 24576
  float* brbuf   = abuf + 24576;           // 24576
  float* pfeat   = brbuf + 24576;          // 73728
  float* protb   = pfeat + 73728;          // 24576
  float* catb    = protb + 24576;          // 49152
  float* gateb   = catb + 49152;           // 24576
  float* finb    = gateb + 24576;          // 49152
  float* fusedb  = finb + 49152;           // 24576
  float* hcat    = fusedb + 24576;         // 49152
  float* h1b     = hcat + 49152;           // 262144
  float* h2b     = h1b + 262144;           // 65536
  __bf16* arena  = (__bf16*)(h2b + 65536); // ~1.7M bf16: transposed padded weights

  // ---- Walk flat params (jax pytree flatten order) + repack weights ----
  size_t off = 0;
  auto take = [&](size_t n) { const float* q = P + off; off += n; return q; };
  auto repackT = [&](const float* src, int K, int N) {
    int Kp = (K + 31) & ~31;
    __bf16* dst = arena; arena += (long)N * Kp;
    k_repackT<<<512, 256, 0, stream>>>(src, dst, K, N, Kp);
    return (const __bf16*)dst;
  };
  auto takeGc = [&](int cin, int cout) {
    Gcbn g;
    const float* Wr = take((size_t)cin * cout);
    const float* Ws = take((size_t)cin * cout);
    g.b = take(cout); g.be = take(cout); g.g = take(cout);
    g.Kp = (cin + 31) & ~31;
    g.WrT = repackT(Wr, cin, cout);
    g.WsT = repackT(Ws, cin, cout);
    g.cout = cout;
    return g;
  };

  const float* W1 = take(192 * 1024);
  const float* W2 = take(1024 * 256);
  const float* W3p = take(256 * 1);
  Enc ea; ea.nl = 8;
  for (int li = 0; li < 8; ++li) { ea.b1c1[li] = takeGc(32 + li * 32, 64);  ea.b1c2[li] = takeGc(64, 32); }
  for (int li = 0; li < 8; ++li) { ea.b2c1[li] = takeGc(144 + li * 32, 64); ea.b2c2[li] = takeGc(64, 32); }
  { const float* cW = take(200 * 96); ea.cb = take(96); ea.cKp = 224; ea.cWT = repackT(cW, 200, 96); }
  ea.conv0 = takeGc(22, 32);
  ea.trans1 = takeGc(288, 144);
  ea.trans2 = takeGc(400, 200);
  const float* b1 = take(1024);
  const float* b2 = take(256);
  const float* b3 = take(1);
  Enc eb; eb.nl = 4;
  for (int li = 0; li < 4; ++li) { eb.b1c1[li] = takeGc(32 + li * 32, 32); eb.b1c2[li] = takeGc(32, 32); }
  for (int li = 0; li < 4; ++li) { eb.b2c1[li] = takeGc(80 + li * 32, 32); eb.b2c2[li] = takeGc(32, 32); }
  { const float* cW = take(104 * 96); eb.cb = take(96); eb.cKp = 128; eb.cWT = repackT(cW, 104, 96); }
  eb.conv0 = takeGc(768, 32);
  eb.trans1 = takeGc(160, 80);
  eb.trans2 = takeGc(208, 104);
  const float* fWs = take(192 * 96); const float* fb = take(96);
  const float* gWs = take(192 * 96); const float* gb = take(96);
  const __bf16* fWT = repackT(fWs, 192, 96);
  const __bf16* gWT = repackT(gWs, 192, 96);
  struct CL { const float* b; const __bf16* wT; int cin; } pl[3][3];
  for (int i = 0; i < 3; ++i)
    for (int j = 0; j <= i; ++j) {
      int cin = (j == 0) ? 128 : 96;
      pl[i][j].b = take(96);
      const float* w = take((size_t)96 * cin * 3);
      __bf16* wT = arena; arena += 96L * 3 * cin;
      k_repack_convT<<<512, 256, 0, stream>>>(w, wT, cin);
      pl[i][j].wT = wT;
      pl[i][j].cin = cin;
    }
  const float* embedW = take(21 * 128);
  const float* lWs = take(288 * 96);
  const float* lb = take(96);
  const __bf16* lWT = repackT(lWs, 288, 96);
  const __bf16* W1T = repackT(W1, 192, 1024);
  const __bf16* W2T = repackT(W2, 1024, 256);
  const __bf16* W3T = repackT(W3p, 256, 1);

  // =========================== Protein tower ===========================
  float* pA = AX; float* pB = AG;
  long padElems = Bn * (L + 2) * 128;
  k_zero<<<4096, 256, 0, stream>>>(pA, padElems);
  k_zero<<<4096, 256, 0, stream>>>(pB, padElems);
  for (int blk = 0; blk < 3; ++blk) {
    k_embed<<<4096, 256, 0, stream>>>(target, embedW, pA, (int)Bn, (int)L);
    float* cur = pA; float* nxt = pB;
    for (int j = 0; j <= blk; ++j) {
      dim3 g(96 / 32, (unsigned)((Bn * L + 15) / 16));
      k_conv_gemm<<<g, 32, 0, stream>>>(cur, pl[blk][j].wT, pl[blk][j].b, nxt, (int)Bn, (int)L, pl[blk][j].cin);
      float* t = cur; cur = nxt; nxt = t;
    }
    k_maxpool<<<(unsigned)((Bn * 96 + 255) / 256), 256, 0, stream>>>(cur, (int)Bn, (int)L, pfeat, 288, blk * 96);
  }
  gemm_f(stream, 0, 0, pfeat, 288, lWT, 288, lb, protb, 96, (int)Bn, 96, 288);

  // =========================== Graph encoders ==========================
  k_pad_copy<<<2048, 256, 0, stream>>>(atomic_x, 22, TH, 32, NA);
  run_encoder(stream, TH, 32, NA, asrc, adst, EA, abid, ea,
              AX, 512, AG, 512, AH, 256, TH, 64, stats, poolsum, poolcnt, abuf);
  run_encoder(stream, brics_x, 768, NB, bsrc, bdst, EB, bbid, eb,
              BX, 256, BG, 768, BHh, 128, TH, 64, stats, poolsum, poolcnt, brbuf);

  // ============================ Fusion MLP =============================
  k_concat2<<<2048, 256, 0, stream>>>(abuf, brbuf, catb, (int)Bn, 96);
  gemm_f(stream, 0, 2, catb, 192, gWT, 192, gb, gateb, 96, (int)Bn, 96, 192);
  k_gate_fuse<<<96, 256, 0, stream>>>(gateb, abuf, brbuf, finb, (int)Bn);
  gemm_f(stream, 0, 0, finb, 192, fWT, 192, fb, fusedb, 96, (int)Bn, 96, 192);
  k_concat2<<<2048, 256, 0, stream>>>(protb, fusedb, hcat, (int)Bn, 96);
  gemm_f(stream, 0, 1, hcat, 192, W1T, 192, b1, h1b, 1024, (int)Bn, 1024, 192);
  gemm_f(stream, 0, 1, h1b, 1024, W2T, 1024, b2, h2b, 256, (int)Bn, 256, 1024);
  gemm_f(stream, 0, 0, h2b, 256, W3T, 256, b3, (float*)d_out, 1, (int)Bn, 1, 256);
}
